// GraphSAGE_35699768164805
// MI455X (gfx1250) — matrix operand order repaired and checked by reading, not compiled, not run on hardware
//
#include <hip/hip_runtime.h>

// ---------------------------------------------------------------------------
// GraphSAGE 2-layer forward for MI455X (gfx1250, wave32, WMMA).
//   N=50000 nodes, E=800000 edges, IN=256, HID=512, OUT=256.
// Strategy:
//   - scatter/mean-aggregate with f32 atomics (L2-resident, traffic-bound)
//   - aggregate scaled+converted to bf16 in one streaming pass
//   - GEMMs on v_wmma_f32_16x16x32_bf16; A panels staged in LDS per block
//     (async global->LDS when available), B (weights) streamed from L2
//   - hidden activations kept in bf16 to halve layer-2 gather bytes
// Workspace: ~232 MB (inv_deg, xb, hb, agg f32, aggb bf16, bf16 weights).
// ---------------------------------------------------------------------------

typedef __attribute__((ext_vector_type(16))) __bf16 v16bf;
typedef __attribute__((ext_vector_type(8)))  float  v8f;
typedef __attribute__((ext_vector_type(4)))  int    v4i;

#define AS_GLOBAL __attribute__((address_space(1)))
#define AS_LDS    __attribute__((address_space(3)))

#if defined(__has_builtin)
#if __has_builtin(__builtin_amdgcn_global_load_async_to_lds_b128)
#define USE_ASYNC_LDS 1
#endif
#if __has_builtin(__builtin_amdgcn_s_wait_asynccnt)
#define HAVE_WAIT_ASYNC_BUILTIN 1
#endif
#endif

__device__ __forceinline__ void wait_asynccnt0() {
#if defined(HAVE_WAIT_ASYNC_BUILTIN)
    __builtin_amdgcn_s_wait_asynccnt(0);
#else
    asm volatile("s_wait_asynccnt 0" ::: "memory");
#endif
}

union FragBF {
    v16bf          v;
    unsigned short u[16];
    uint4          q[2];
};

__device__ __forceinline__ unsigned short f2bf(float f) {
    unsigned int u = __float_as_uint(f);
    u += 0x7FFFu + ((u >> 16) & 1u);        // round-to-nearest-even
    return (unsigned short)(u >> 16);
}
__device__ __forceinline__ unsigned int f2bf_pk(float lo, float hi) {
    return (unsigned int)f2bf(lo) | ((unsigned int)f2bf(hi) << 16);
}
__device__ __forceinline__ float bf2f(unsigned int h) {
    return __uint_as_float(h << 16);
}

// ------------------------------ utility kernels ----------------------------

__global__ void k_zero4(float4* __restrict__ p, long long n4) {
    long long i = (long long)blockIdx.x * blockDim.x + threadIdx.x;
    if (i < n4) p[i] = make_float4(0.f, 0.f, 0.f, 0.f);
}

__global__ void k_cvt_bf16x8(const float* __restrict__ in,
                             unsigned short* __restrict__ out, long long n8) {
    long long i = (long long)blockIdx.x * blockDim.x + threadIdx.x;
    if (i >= n8) return;
    float4 a = ((const float4*)in)[i * 2];
    float4 b = ((const float4*)in)[i * 2 + 1];
    uint4 o;
    o.x = f2bf_pk(a.x, a.y);
    o.y = f2bf_pk(a.z, a.w);
    o.z = f2bf_pk(b.x, b.y);
    o.w = f2bf_pk(b.z, b.w);
    ((uint4*)out)[i] = o;
}

// aggb[r,c] = bf16( agg[r,c] * inv_deg[r] ), 8 elements per thread
template <int C>
__global__ void k_scale_cvt_bf16(const float* __restrict__ in,
                                 const float* __restrict__ inv_deg,
                                 unsigned short* __restrict__ out, int Nrows) {
    constexpr int PER = C >> 3;
    long long i = (long long)blockIdx.x * blockDim.x + threadIdx.x;
    if (i >= (long long)Nrows * PER) return;
    int r = (int)(i / PER);
    int c = (int)(i % PER) << 3;
    float s = inv_deg[r];
    const float4* p = (const float4*)(in + (size_t)r * C + c);
    float4 a = p[0];
    float4 b = p[1];
    uint4 o;
    o.x = f2bf_pk(a.x * s, a.y * s);
    o.y = f2bf_pk(a.z * s, a.w * s);
    o.z = f2bf_pk(b.x * s, b.y * s);
    o.w = f2bf_pk(b.z * s, b.w * s);
    *(uint4*)(out + (size_t)r * C + c) = o;
}

__global__ void k_degree(const int* __restrict__ dst, float* __restrict__ deg, int E) {
    int e = blockIdx.x * blockDim.x + threadIdx.x;
    if (e < E) atomicAdd(&deg[dst[e]], 1.0f);
}

__global__ void k_invdeg(float* __restrict__ d, int n) {
    int i = blockIdx.x * blockDim.x + threadIdx.x;
    if (i < n) d[i] = 1.0f / fmaxf(d[i], 1.0f);
}

// ------------------------------ scatter kernel -----------------------------

// agg[dst] += Xb[src]  (bf16 source gather, f32 atomic accumulate; 8 cols/thread)
template <int C>
__global__ void k_scatter_bf16(const unsigned short* __restrict__ Xb,
                               const int* __restrict__ src,
                               const int* __restrict__ dst,
                               float* __restrict__ agg, int E) {
    constexpr int PER = C >> 3;
    long long t = (long long)blockIdx.x * blockDim.x + threadIdx.x;
    if (t >= (long long)E * PER) return;
    int e = (int)(t / PER);
    int c = (int)(t % PER) << 3;
    int s = src[e], d = dst[e];
    uint4 q = *(const uint4*)(Xb + (size_t)s * C + c);
    float* o = agg + (size_t)d * C + c;
    atomicAdd(o + 0, bf2f(q.x & 0xFFFFu));
    atomicAdd(o + 1, bf2f(q.x >> 16));
    atomicAdd(o + 2, bf2f(q.y & 0xFFFFu));
    atomicAdd(o + 3, bf2f(q.y >> 16));
    atomicAdd(o + 4, bf2f(q.z & 0xFFFFu));
    atomicAdd(o + 5, bf2f(q.z >> 16));
    atomicAdd(o + 6, bf2f(q.w & 0xFFFFu));
    atomicAdd(o + 7, bf2f(q.w >> 16));
}

// ------------------------------ fused WMMA GEMM ----------------------------
// out[N, NCOL] = act( Aagg @ Wl^T + Ax @ Wr^T + bias )
//   Aagg : [N, K] bf16 (pre-scaled by 1/deg);  Ax : [N, K] bf16
//   Wl,Wr: [NCOL, K] bf16 row-major (B operand: two runs of 8 bf16 per lane, 16 apart)
// The 16xK A panels (shared by all 4 waves of the block) are staged in LDS
// once (async global->LDS when available); the k-loop reads A via ds_load_b128
// and streams only the L2-resident weights from global memory.
// Each wave computes a 16x64 strip: 4 accumulator tiles.
// Block = 128 threads = 4 waves; NCOL/256 blocks per 16-row tile.
template <int K, int NCOL, bool RELU, bool OUT_BF16>
__global__ __launch_bounds__(128)
void k_sage_gemm(const unsigned short* __restrict__ Aagg,
                 const unsigned short* __restrict__ Ax,
                 const unsigned short* __restrict__ Wl,
                 const unsigned short* __restrict__ Wr,
                 const float* __restrict__ bias,
                 void* __restrict__ outp, int Nrows) {
    constexpr int BPR  = NCOL / 256;      // blocks per 16-row tile
    constexpr int PAD  = 8;               // bf16 elems: 16B row pad -> LDS row
    constexpr int LROW = K + PAD;         //   stride = 4 dwords mod 64 banks
    __shared__ unsigned short As[16 * LROW];
    __shared__ unsigned short Xs[16 * LROW];

    const int wave = threadIdx.x >> 5;
    const int lane = threadIdx.x & 31;
    const int lm   = lane & 15;           // lane-in-half
    const int lhi  = lane >> 4;           // which K half
    const int row0 = (blockIdx.x / BPR) * 16;
    if (row0 >= Nrows) return;
    const int n0 = ((blockIdx.x % BPR) * 4 + wave) * 64;

    // ---- stage both 16xK A panels into LDS (once per block) ----
    constexpr int CPR  = K / 8;           // uint4 chunks per row
    constexpr int PERT = (16 * CPR) / 128;// chunks per thread per panel (4 or 8)
#pragma unroll
    for (int i = 0; i < PERT; ++i) {
        const int c   = i * 128 + threadIdx.x;
        const int r   = c / CPR;
        const int col = (c % CPR) * 8;
        const unsigned short* gA = Aagg + (size_t)(row0 + r) * K + col;
        const unsigned short* gX = Ax   + (size_t)(row0 + r) * K + col;
        unsigned short* lA = &As[r * LROW + col];
        unsigned short* lX = &Xs[r * LROW + col];
#if defined(USE_ASYNC_LDS)
        __builtin_amdgcn_global_load_async_to_lds_b128(
            (AS_GLOBAL v4i*)gA, (AS_LDS v4i*)lA, 0, 0);
        __builtin_amdgcn_global_load_async_to_lds_b128(
            (AS_GLOBAL v4i*)gX, (AS_LDS v4i*)lX, 0, 0);
#else
        *(uint4*)lA = *(const uint4*)gA;
        *(uint4*)lX = *(const uint4*)gX;
#endif
    }
#if defined(USE_ASYNC_LDS)
    wait_asynccnt0();                     // own async writes visible in LDS
#endif
    __syncthreads();

    v8f acc[4] = {v8f{}, v8f{}, v8f{}, v8f{}};

    // A layout: lanes 0-15 hold K=k..k+7 & k+16..k+23 of row lm;
    //           lanes 16-31 hold K=k+8..k+15 & k+24..k+31 of row lm.
    const unsigned short* arow = &As[lm * LROW + (lhi << 3)];
    const unsigned short* xrow = &Xs[lm * LROW + (lhi << 3)];

    for (int k0 = 0; k0 < K; k0 += 32) {
        FragBF fa, fx;
        fa.q[0] = *(const uint4*)(arow + k0);       // ds_load_b128
        fa.q[1] = *(const uint4*)(arow + k0 + 16);
        fx.q[0] = *(const uint4*)(xrow + k0);
        fx.q[1] = *(const uint4*)(xrow + k0 + 16);

        // 4 column tiles, two WMMAs each (Wl on agg, Wr on x)
#pragma unroll
        for (int j = 0; j < 4; ++j) {
            const int n = n0 + j * 16 + lm;
            // B layout: lanes 0-15 col n K=k..k+7 & k+16..k+23; lanes 16-31 col n K=k+8..k+15 & k+24..k+31
            const unsigned short* pl = Wl + (size_t)n * K + k0 + (lhi << 3);
            const unsigned short* pr = Wr + (size_t)n * K + k0 + (lhi << 3);
            FragBF fb, fr;
            fb.q[0] = *(const uint4*)pl;  fb.q[1] = *(const uint4*)(pl + 16);
            fr.q[0] = *(const uint4*)pr;  fr.q[1] = *(const uint4*)(pr + 16);
            acc[j] = __builtin_amdgcn_wmma_f32_16x16x32_bf16(
                false, fa.v, false, fb.v, (short)0, acc[j], false, false);
            acc[j] = __builtin_amdgcn_wmma_f32_16x16x32_bf16(
                false, fx.v, false, fr.v, (short)0, acc[j], false, false);
        }
    }

    // Epilogue. C/D layout: lanes 0-15 rows 0-7 in v0..7, lanes 16-31 rows 8-15.
    const int mbase = row0 + (lhi << 3);
#pragma unroll
    for (int j = 0; j < 4; ++j) {
        const int n = n0 + j * 16 + lm;
        const float b = bias[n];
#pragma unroll
        for (int v = 0; v < 8; ++v) {
            float val = acc[j][v] + b;
            if (RELU) val = fmaxf(val, 0.0f);
            if (OUT_BF16)
                ((unsigned short*)outp)[(size_t)(mbase + v) * NCOL + n] = f2bf(val);
            else
                ((float*)outp)[(size_t)(mbase + v) * NCOL + n] = val;
        }
    }
}

// ------------------------------ host launcher ------------------------------

extern "C" void kernel_launch(void* const* d_in, const int* in_sizes, int n_in,
                              void* d_out, int out_size, void* d_ws, size_t ws_size,
                              hipStream_t stream) {
    constexpr int IN = 256, HID = 512, OUT = 256;
    const int N = in_sizes[0] / IN;
    const int E = in_sizes[1] / 2;

    const float* x   = (const float*)d_in[0];
    const int*   ei  = (const int*)d_in[1];     // int32 (JAX default / harness rule)
    const float* W1l = (const float*)d_in[2];
    const float* b1  = (const float*)d_in[3];
    const float* W1r = (const float*)d_in[4];
    const float* W2l = (const float*)d_in[5];
    const float* b2  = (const float*)d_in[6];
    const float* W2r = (const float*)d_in[7];
    const int* srcI = ei;
    const int* dstI = ei + E;

    // ---- workspace carve-out (~232 MB) ----
    char*  ws  = (char*)d_ws;
    size_t off = 0;
    auto take = [&](size_t bytes) -> char* {
        char* p = ws + off;
        off = (off + bytes + 255) & ~(size_t)255;
        return p;
    };
    float*          inv_deg = (float*)take((size_t)N * 4);
    unsigned short* xb      = (unsigned short*)take((size_t)N * IN * 2);
    unsigned short* hb      = (unsigned short*)take((size_t)N * HID * 2);
    float*          agg     = (float*)take((size_t)N * HID * 4);          // reused both layers
    unsigned short* aggb    = (unsigned short*)take((size_t)N * HID * 2); // pre-scaled bf16
    unsigned short* w1l_b   = (unsigned short*)take((size_t)HID * IN * 2);
    unsigned short* w1r_b   = (unsigned short*)take((size_t)HID * IN * 2);
    unsigned short* w2l_b   = (unsigned short*)take((size_t)OUT * HID * 2);
    unsigned short* w2r_b   = (unsigned short*)take((size_t)OUT * HID * 2);

    const int TB = 256;
    auto blocks = [&](long long work) { return (unsigned)((work + TB - 1) / TB); };

    // 1) degrees -> inv_deg
    k_zero4<<<blocks((long long)N / 4), TB, 0, stream>>>((float4*)inv_deg, (long long)N / 4);
    k_degree<<<blocks(E), TB, 0, stream>>>(dstI, inv_deg, E);
    k_invdeg<<<blocks(N), TB, 0, stream>>>(inv_deg, N);

    // 2) bf16 conversions (x and all weights)
    k_cvt_bf16x8<<<blocks((long long)N * IN / 8), TB, 0, stream>>>(x, xb, (long long)N * IN / 8);
    k_cvt_bf16x8<<<blocks((long long)HID * IN / 8), TB, 0, stream>>>(W1l, w1l_b, (long long)HID * IN / 8);
    k_cvt_bf16x8<<<blocks((long long)HID * IN / 8), TB, 0, stream>>>(W1r, w1r_b, (long long)HID * IN / 8);
    k_cvt_bf16x8<<<blocks((long long)OUT * HID / 8), TB, 0, stream>>>(W2l, w2l_b, (long long)OUT * HID / 8);
    k_cvt_bf16x8<<<blocks((long long)OUT * HID / 8), TB, 0, stream>>>(W2r, w2r_b, (long long)OUT * HID / 8);

    // 3) layer-1 aggregate: agg[0:N,0:256] = sum_{e} x[src]  (bf16 gather)
    k_zero4<<<blocks((long long)N * IN / 4), TB, 0, stream>>>((float4*)agg, (long long)N * IN / 4);
    k_scatter_bf16<IN><<<blocks((long long)E * (IN / 8)), TB, 0, stream>>>(xb, srcI, dstI, agg, E);
    k_scale_cvt_bf16<IN><<<blocks((long long)N * (IN / 8)), TB, 0, stream>>>(agg, inv_deg, aggb, N);

    // 4) layer-1 fused GEMM: hb = relu(aggb @ W1l^T + xb @ W1r^T + b1)  [bf16]
    {
        const int mtiles = (N + 15) / 16;
        dim3 grid(mtiles * (HID / 256));
        k_sage_gemm<IN, HID, true, true><<<grid, 128, 0, stream>>>(
            aggb, xb, w1l_b, w1r_b, b1, hb, N);
    }

    // 5) layer-2 aggregate: agg[0:N,0:512] = sum_{e} h[src]  (bf16 gather, f32 atomics)
    k_zero4<<<blocks((long long)N * HID / 4), TB, 0, stream>>>((float4*)agg, (long long)N * HID / 4);
    k_scatter_bf16<HID><<<blocks((long long)E * (HID / 8)), TB, 0, stream>>>(hb, srcI, dstI, agg, E);
    k_scale_cvt_bf16<HID><<<blocks((long long)N * (HID / 8)), TB, 0, stream>>>(agg, inv_deg, aggb, N);

    // 6) layer-2 fused GEMM: z = aggb @ W2l^T + hb @ W2r^T + b2  [f32 -> d_out]
    {
        const int mtiles = (N + 15) / 16;
        dim3 grid(mtiles * (OUT / 256));
        k_sage_gemm<HID, OUT, false, false><<<grid, 128, 0, stream>>>(
            aggb, hb, w2l_b, w2r_b, b2, d_out, N);
    }
}
